// VarianceAdaptor_34789235098001
// MI455X (gfx1250) — compile-verified, matrix operand order checked by reading, not run on hardware
//
#include <hip/hip_runtime.h>
#include <hip/hip_bf16.h>

typedef __attribute__((ext_vector_type(16))) __bf16 v16bf;
typedef __attribute__((ext_vector_type(8)))  __bf16 v8bf;
typedef __attribute__((ext_vector_type(8)))  float  v8f;
typedef __attribute__((ext_vector_type(4)))  float  v4f;

#define BATCH 32
#define TSEQ  256
#define DIM   384
#define TARGET 2048
#define MROWS (BATCH*TSEQ)          // 8192
#define WELEMS (3*DIM*DIM)          // 442368

// ---------------------------------------------------------------------------
// Pack weights: (k, d, n) f32 -> (k, n, d) bf16  (B-fragment friendly layout)
__global__ __launch_bounds__(256) void pack_w_kernel(const float* __restrict__ w,
                                                     __bf16* __restrict__ wB) {
  int id = blockIdx.x * 256 + threadIdx.x;
  if (id >= WELEMS) return;
  int k = id / (DIM * DIM);
  int r = id - k * DIM * DIM;
  int d = r / DIM;
  int n = r - d * DIM;
  wB[((size_t)(k * DIM + n)) * DIM + d] = (__bf16)w[id];
}

// Convert activations f32 -> bf16 flat
__global__ __launch_bounds__(256) void pack_x_kernel(const float* __restrict__ x,
                                                     __bf16* __restrict__ xb, int n) {
  int id = blockIdx.x * 256 + threadIdx.x;
  if (id < n) xb[id] = (__bf16)x[id];
}

// ---------------------------------------------------------------------------
// Length regulation step 1: per-batch cumsum + searchsorted -> gather indices
__global__ __launch_bounds__(256) void reg_scan_kernel(const float* __restrict__ dur,
                                                       int* __restrict__ gidx,
                                                       float* __restrict__ maskOut) {
  __shared__ int cum[TSEQ];
  const int tid = threadIdx.x;
  const int b = blockIdx.x;
  float dv = dur[b * TSEQ + tid];
  dv = fminf(fmaxf(dv, 1.0f), 8.0f);
  cum[tid] = (int)rintf(dv);             // round-half-even, matches jnp.round
  __syncthreads();
  for (int off = 1; off < TSEQ; off <<= 1) {
    int v = (tid >= off) ? cum[tid - off] : 0;
    __syncthreads();
    cum[tid] += v;
    __syncthreads();
  }
  const int total = cum[TSEQ - 1];
  for (int t = tid; t < TARGET; t += 256) {
    int lo = 0, hi = TSEQ;                // searchsorted(cum, t, 'right')
    while (lo < hi) { int mid = (lo + hi) >> 1; if (cum[mid] <= t) lo = mid + 1; else hi = mid; }
    bool valid = t < total;
    int gi = lo < (TSEQ - 1) ? lo : (TSEQ - 1);
    gidx[b * TARGET + t] = valid ? gi : -1;
    maskOut[b * TARGET + t] = valid ? 1.0f : 0.0f;
  }
}

// Length regulation step 2: pure-bandwidth float4 gather (dominant HBM traffic)
__global__ __launch_bounds__(256) void reg_gather_kernel(const float* __restrict__ enc,
                                                         const int* __restrict__ gidx,
                                                         float* __restrict__ outReg) {
  int id = blockIdx.x * 256 + threadIdx.x;      // B*TARGET*(DIM/4) threads
  int b = id / (TARGET * (DIM / 4));
  int rem = id - b * (TARGET * (DIM / 4));
  int t = rem / (DIM / 4);
  int c = rem - t * (DIM / 4);
  int g = gidx[b * TARGET + t];
  v4f v = {};
  if (g >= 0) v = *(const v4f*)(enc + ((size_t)b * TSEQ + g) * DIM + c * 4);
  *(v4f*)(outReg + ((size_t)b * TARGET + t) * DIM + c * 4) = v;
}

// ---------------------------------------------------------------------------
// Conv1d (K=3, SAME) as WMMA GEMM:
//   Y[m,n] = relu( sum_tap sum_d X[m+tap-1,d]*W[tap,d,n] + bias[n] )
// X staged in LDS (bf16, padded stride); W pre-transposed to (tap, n, d) bf16.
// Each wave owns one 16-wide N-tile and 4 M-tiles. Per k-step: one B fragment
// from global (software-pipelined one step ahead) feeds 4 independent WMMAs;
// all 4 M-tiles' A fragments are staged in registers first (8 ds_load_b128 in
// one clause) so the WMMA quartet retires with staggered partial dscnt waits.
#define MTILE 64
#define LDS_STRIDE 392   // 384 + 8 halves pad -> b128 lanes land on distinct banks
#define KSTEPS 36        // 3 taps * 12 k-tiles of 32
__global__ __launch_bounds__(256)
void conv_gemm_kernel(const __bf16* __restrict__ Xb, const __bf16* __restrict__ Wb,
                      const float* __restrict__ bias, float* __restrict__ Y) {
  __shared__ __attribute__((aligned(16))) __bf16 As[(MTILE + 2) * LDS_STRIDE];
  const int tid = threadIdx.x;
  const int m0 = blockIdx.x * MTILE;
  const int bStart = m0 & ~(TSEQ - 1);          // batch boundary for SAME zero-pad

  // Stage rows m0-1 .. m0+MTILE into LDS (bf16, 8-half chunks, coalesced)
  for (int c = tid; c < (MTILE + 2) * (DIM / 8); c += 256) {
    int r = c / (DIM / 8);
    int cc = (c - r * (DIM / 8)) * 8;
    int g = m0 - 1 + r;
    v8bf val = {};
    if (g >= bStart && g < bStart + TSEQ)
      val = *(const v8bf*)(Xb + (size_t)g * DIM + cc);
    *(v8bf*)(As + r * LDS_STRIDE + cc) = val;
  }
  __syncthreads();

  const int wave = tid >> 5, lane = tid & 31;
  const int lm = lane & 15, lh = lane >> 4;
  const int n0 = (blockIdx.y * 8 + wave) * 16;  // gridDim.y == 3 covers N=384

  v8f acc[4] = {{}, {}, {}, {}};
  // B column base for this lane: (tap, n0+lm, d) ; step s -> tap=s/12, kt=s%12
  const __bf16* bcol = Wb + (size_t)(n0 + lm) * DIM + lh * 8;

  v8bf blo = *(const v8bf*)(bcol);
  v8bf bhi = *(const v8bf*)(bcol + 16);
#pragma unroll
  for (int s = 0; s < KSTEPS; ++s) {
    const int tap = s / 12, kt = s - tap * 12;
    // Prefetch next step's B fragment (global) before issuing this step's WMMAs
    v8bf nlo = {}, nhi = {};
    if (s + 1 < KSTEPS) {
      const int tap2 = (s + 1) / 12, kt2 = (s + 1) - tap2 * 12;
      const __bf16* nb = bcol + (size_t)tap2 * DIM * DIM + kt2 * 32;
      nlo = *(const v8bf*)nb;
      nhi = *(const v8bf*)(nb + 16);
    }
    v16bf Bf = __builtin_shufflevector(blo, bhi, 0,1,2,3,4,5,6,7,8,9,10,11,12,13,14,15);
    const __bf16* acol = As + (size_t)(tap + lm) * LDS_STRIDE + kt * 32 + lh * 8;
    // Stage all 4 M-tiles' A fragments (one ds clause), then issue 4 WMMAs
    v8bf alo[4], ahi[4];
#pragma unroll
    for (int mt = 0; mt < 4; ++mt) {
      alo[mt] = *(const v8bf*)(acol + mt * 16 * LDS_STRIDE);
      ahi[mt] = *(const v8bf*)(acol + mt * 16 * LDS_STRIDE + 16);
    }
#pragma unroll
    for (int mt = 0; mt < 4; ++mt) {
      v16bf Af = __builtin_shufflevector(alo[mt], ahi[mt],
                                         0,1,2,3,4,5,6,7,8,9,10,11,12,13,14,15);
      acc[mt] = __builtin_amdgcn_wmma_f32_16x16x32_bf16(false, Af, false, Bf,
                                                        (short)0, acc[mt], false, false);
    }
    blo = nlo; bhi = nhi;
  }

  // Epilogue: bias + relu.  D layout: lane holds n = n0+lm, rows m = lh*8 + v
  const float bn = bias[n0 + lm];
#pragma unroll
  for (int mt = 0; mt < 4; ++mt) {
#pragma unroll
    for (int v = 0; v < 8; ++v) {
      float val = acc[mt][v] + bn;
      val = val > 0.0f ? val : 0.0f;
      Y[(size_t)(m0 + mt * 16 + lh * 8 + v) * DIM + n0 + lm] = val;
    }
  }
}

// ---------------------------------------------------------------------------
// Block reduction: wave32 shuffle reduce + 4-entry LDS combine (2 barriers)
__device__ __forceinline__ float block_sum(float v, float* red, int tid) {
#pragma unroll
  for (int off = 16; off > 0; off >>= 1) v += __shfl_xor(v, off, 32);
  if ((tid & 31) == 0) red[tid >> 5] = v;
  __syncthreads();
  float s = red[0] + red[1] + red[2] + red[3];
  __syncthreads();                               // red reusable after this
  return s;
}

// LayerNorm(residual + y) -> f32 (+ bf16 copy for next GEMM)
__global__ __launch_bounds__(128)
void ln_kernel(const float* __restrict__ Yin, const float* __restrict__ Xres,
               const float* __restrict__ g, const float* __restrict__ be,
               float* __restrict__ outF, __bf16* __restrict__ outB) {
  __shared__ float red[4];
  const int tid = threadIdx.x;
  const size_t row = blockIdx.x;
  const float* yr = Yin + row * DIM;
  const float* xr = Xres + row * DIM;
  float x0 = yr[tid]       + xr[tid];
  float x1 = yr[tid + 128] + xr[tid + 128];
  float x2 = yr[tid + 256] + xr[tid + 256];
  float mean = block_sum(x0 + x1 + x2, red, tid) * (1.0f / DIM);
  float d0 = x0 - mean, d1 = x1 - mean, d2 = x2 - mean;
  float var = block_sum(d0 * d0 + d1 * d1 + d2 * d2, red, tid) * (1.0f / DIM);
  float rs = rsqrtf(var + 1e-6f);
  float y0 = d0 * rs * g[tid]       + be[tid];
  float y1 = d1 * rs * g[tid + 128] + be[tid + 128];
  float y2 = d2 * rs * g[tid + 256] + be[tid + 256];
  outF[row * DIM + tid] = y0;       outB[row * DIM + tid] = (__bf16)y0;
  outF[row * DIM + tid + 128] = y1; outB[row * DIM + tid + 128] = (__bf16)y1;
  outF[row * DIM + tid + 256] = y2; outB[row * DIM + tid + 256] = (__bf16)y2;
}

// LayerNorm + fused duration projection: dur[row] = dot(LN(y+res), wd) + bd
__global__ __launch_bounds__(128)
void ln_dur_kernel(const float* __restrict__ Yin, const float* __restrict__ Xres,
                   const float* __restrict__ g, const float* __restrict__ be,
                   const float* __restrict__ wd, const float* __restrict__ bd,
                   float* __restrict__ durOut) {
  __shared__ float red[4];
  const int tid = threadIdx.x;
  const size_t row = blockIdx.x;
  const float* yr = Yin + row * DIM;
  const float* xr = Xres + row * DIM;
  float x0 = yr[tid]       + xr[tid];
  float x1 = yr[tid + 128] + xr[tid + 128];
  float x2 = yr[tid + 256] + xr[tid + 256];
  float mean = block_sum(x0 + x1 + x2, red, tid) * (1.0f / DIM);
  float d0 = x0 - mean, d1 = x1 - mean, d2 = x2 - mean;
  float var = block_sum(d0 * d0 + d1 * d1 + d2 * d2, red, tid) * (1.0f / DIM);
  float rs = rsqrtf(var + 1e-6f);
  float y0 = d0 * rs * g[tid]       + be[tid];
  float y1 = d1 * rs * g[tid + 128] + be[tid + 128];
  float y2 = d2 * rs * g[tid + 256] + be[tid + 256];
  float dd = block_sum(y0 * wd[tid] + y1 * wd[tid + 128] + y2 * wd[tid + 256], red, tid);
  if (tid == 0) durOut[row] = dd + bd[0];
}

// ---------------------------------------------------------------------------
extern "C" void kernel_launch(void* const* d_in, const int* in_sizes, int n_in,
                              void* d_out, int out_size, void* d_ws, size_t ws_size,
                              hipStream_t stream) {
  const float* enc = (const float*)d_in[0];
  const float* dur = (const float*)d_in[1];
  const float* w1  = (const float*)d_in[2];
  const float* b1  = (const float*)d_in[3];
  const float* w2  = (const float*)d_in[4];
  const float* b2  = (const float*)d_in[5];
  const float* g1  = (const float*)d_in[6];
  const float* be1 = (const float*)d_in[7];
  const float* g2  = (const float*)d_in[8];
  const float* be2 = (const float*)d_in[9];
  const float* wd  = (const float*)d_in[10];
  const float* bd  = (const float*)d_in[11];
  (void)in_sizes; (void)n_in; (void)out_size; (void)ws_size;

  // Workspace layout (all 16B-aligned offsets)
  char* ws = (char*)d_ws;
  __bf16* wB1  = (__bf16*)(ws + 0);                        //   884,736 B
  __bf16* wB2  = (__bf16*)(ws + 884736);                   //   884,736 B
  __bf16* encB = (__bf16*)(ws + 1769472);                  // 6,291,456 B
  float*  Ybuf = (float*) (ws + 8060928);                  // 12,582,912 B (reused by both convs)
  float*  c1f  = (float*) (ws + 20643840);                 // 12,582,912 B
  __bf16* c1B  = (__bf16*)(ws + 33226752);                 // 6,291,456 B
  int*    gidx = (int*)   (ws + 39518208);                 //   262,144 B

  float* out     = (float*)d_out;
  float* outReg  = out;                                    // (32,2048,384)
  float* outDur  = out + (size_t)BATCH * TARGET * DIM;     // (32,256)
  float* outMask = outDur + (size_t)BATCH * TSEQ;          // (32,2048)

  // Pack weights and activations to bf16 / fragment-friendly layouts
  pack_w_kernel<<<WELEMS / 256, 256, 0, stream>>>(w1, wB1);
  pack_w_kernel<<<WELEMS / 256, 256, 0, stream>>>(w2, wB2);
  pack_x_kernel<<<(MROWS * DIM) / 256, 256, 0, stream>>>(enc, encB, MROWS * DIM);

  // Length regulation (independent of conv chain)
  reg_scan_kernel<<<BATCH, 256, 0, stream>>>(dur, gidx, outMask);
  reg_gather_kernel<<<(BATCH * TARGET * (DIM / 4)) / 256, 256, 0, stream>>>(enc, gidx, outReg);

  // Conv1 -> relu -> LN(+enc) -> Conv2 -> relu -> LN(+c1) -> dot(wd)
  dim3 ggrid(MROWS / MTILE, 3);
  conv_gemm_kernel<<<ggrid, 256, 0, stream>>>(encB, wB1, b1, Ybuf);
  ln_kernel<<<MROWS, 128, 0, stream>>>(Ybuf, enc, g1, be1, c1f, c1B);
  conv_gemm_kernel<<<ggrid, 256, 0, stream>>>(c1B, wB2, b2, Ybuf);
  ln_dur_kernel<<<MROWS, 128, 0, stream>>>(Ybuf, c1f, g2, be2, wd, bd, outDur);
}
